// GNNLayer_53661321396293
// MI455X (gfx1250) — matrix operand order repaired and checked by reading, not compiled, not run on hardware
//
#include <hip/hip_runtime.h>

#define D_    256
#define DFF_  512
#define EPS_  1e-5f

typedef __attribute__((ext_vector_type(16))) __bf16 v16bf;
typedef __attribute__((ext_vector_type(8)))  __bf16 v8bf;
typedef __attribute__((ext_vector_type(8)))  float  v8f;

// stride (in bf16 elems) for the LDS-resident FFN intermediate
#define STP_  520
// stride (in floats) for LN staging
#define SBP_  260

// ---------- bf16 conversion: native cast -> v_cvt_pk_bf16_f32 (RNE) ----------
static __device__ __forceinline__ __bf16 f2bf(float f) { return (__bf16)f; }

// ---------- A-tile loader: 16x32 bf16 A from row-major fp32 ----------
// lane: m = lane&15, kh = lane>>4 ; element j -> k = (j&7) + 8*kh + 16*(j>>3)
static __device__ __forceinline__ v16bf load_a_tile(const float* src, int ld, int m, int kb, int kh) {
    const float* p0 = src + (size_t)m * ld + kb + 8 * kh;
    const float* p1 = p0 + 16;
    float4 f0 = *(const float4*)(p0);
    float4 f1 = *(const float4*)(p0 + 4);
    float4 f2 = *(const float4*)(p1);
    float4 f3 = *(const float4*)(p1 + 4);
    v16bf a;
    a[0]  = f2bf(f0.x); a[1]  = f2bf(f0.y); a[2]  = f2bf(f0.z); a[3]  = f2bf(f0.w);
    a[4]  = f2bf(f1.x); a[5]  = f2bf(f1.y); a[6]  = f2bf(f1.z); a[7]  = f2bf(f1.w);
    a[8]  = f2bf(f2.x); a[9]  = f2bf(f2.y); a[10] = f2bf(f2.z); a[11] = f2bf(f2.w);
    a[12] = f2bf(f3.x); a[13] = f2bf(f3.y); a[14] = f2bf(f3.z); a[15] = f2bf(f3.w);
    return a;
}

// ---------- A-tile loader from bf16 LDS (no conversion needed) ----------
static __device__ __forceinline__ v16bf load_a_lds(const __bf16* st, int ld, int m, int kb, int kh) {
    const __bf16* p0 = st + (size_t)m * ld + kb + 8 * kh;
    v8bf lo = *(const v8bf*)(p0);
    v8bf hi = *(const v8bf*)(p0 + 16);
    return __builtin_shufflevector(lo, hi, 0,1,2,3,4,5,6,7,8,9,10,11,12,13,14,15);
}

// ---------- zero workspace ----------
__global__ void zero_f32(float* p, long long n) {
    long long i = (long long)blockIdx.x * blockDim.x + threadIdx.x;
    long long stride = (long long)gridDim.x * blockDim.x;
    for (; i < n; i += stride) p[i] = 0.0f;
}

// ---------- pack fp32 weight [K x Nc] into WMMA B-tile bf16 layout ----------
// B layout: lane l holds col n = nt*16 + (l&15); element j -> k = kt*32 + 8*(l>>4) + (j&7) + 16*(j>>3)
__global__ void pack_weight(const float* __restrict__ W, __bf16* __restrict__ out,
                            int K, int Nc) {
    int idx = blockIdx.x * blockDim.x + threadIdx.x;
    int total = (K / 32) * (Nc / 16) * 32;
    if (idx >= total) return;
    int lane = idx & 31;
    int tile = idx >> 5;
    int NT = Nc / 16;
    int kt = tile / NT;
    int nt = tile % NT;
    int n  = nt * 16 + (lane & 15);
    int kh = lane >> 4;
    __bf16* o = out + (size_t)tile * 512 + lane * 16;
#pragma unroll
    for (int j = 0; j < 16; ++j) {
        int k = kt * 32 + 8 * kh + (j & 7) + 16 * (j >> 3);
        o[j] = f2bf(W[(size_t)k * Nc + n]);
    }
}

// ---------- edge scatter: agg[dst] += x[src] ----------
__global__ __launch_bounds__(256) void scatter_add(const float* __restrict__ x,
                                                   const long long* __restrict__ ei,
                                                   float* __restrict__ agg, int E) {
    long long idx = (long long)blockIdx.x * 256 + threadIdx.x;
    if (idx >= (long long)E * 64) return;
    int e = (int)(idx >> 6);
    int c = (int)(idx & 63) << 2;
    long long s = ei[e];
    long long d = ei[(size_t)E + e];
    float4 v = *(const float4*)(x + s * D_ + c);
    float* ap = agg + d * D_ + c;
    atomicAdd(ap + 0, v.x);
    atomicAdd(ap + 1, v.y);
    atomicAdd(ap + 2, v.z);
    atomicAdd(ap + 3, v.w);
}

// ---------- shared LN helper pieces are inlined in kernels (32 rows/block) ----------

// ---------- kernel 3: x2 = agg@Wn + x@Wr + b ; h = LN1(x + x2); 32 rows/block ----------
__global__ __launch_bounds__(128) void gnn_ln1(const float* __restrict__ x,
                                               const float* __restrict__ agg,
                                               const __bf16* __restrict__ Wn_p,
                                               const __bf16* __restrict__ Wr_p,
                                               const float* __restrict__ b_gnn,
                                               const float* __restrict__ g1,
                                               const float* __restrict__ beta1,
                                               float* __restrict__ h, int N) {
    __shared__ float sbuf[32 * SBP_];
    __shared__ float part[32][4];
    __shared__ float part2[32][4];
    __shared__ float smu[32], srs[32];

    const int t    = threadIdx.x;
    const int wave = t >> 5;
    const int lane = t & 31;
    const int kh   = lane >> 4;
    const int nlo  = lane & 15;
    const int row0a = blockIdx.x * 32;
    const int row0b = min(row0a + 16, N - 16);   // tail block: duplicate tile (benign)

    v8f acc0[4] = {};
    v8f acc1[4] = {};
    const float* aggA = agg + (size_t)row0a * D_;
    const float* aggB = agg + (size_t)row0b * D_;
    const float* xA   = x   + (size_t)row0a * D_;
    const float* xB   = x   + (size_t)row0b * D_;

#pragma unroll
    for (int kt = 0; kt < 8; ++kt) {
        v16bf a0 = load_a_tile(aggA, D_, nlo, kt * 32, kh);
        v16bf a1 = load_a_tile(aggB, D_, nlo, kt * 32, kh);
#pragma unroll
        for (int i = 0; i < 4; ++i) {
            int nt = wave * 4 + i;
            v16bf b = *(const v16bf*)(Wn_p + ((size_t)(kt * 16 + nt) * 32 + lane) * 16);
            acc0[i] = __builtin_amdgcn_wmma_f32_16x16x32_bf16(false, a0, false, b,
                                                              (short)0, acc0[i], false, false);
            acc1[i] = __builtin_amdgcn_wmma_f32_16x16x32_bf16(false, a1, false, b,
                                                              (short)0, acc1[i], false, false);
        }
    }
#pragma unroll
    for (int kt = 0; kt < 8; ++kt) {
        v16bf a0 = load_a_tile(xA, D_, nlo, kt * 32, kh);
        v16bf a1 = load_a_tile(xB, D_, nlo, kt * 32, kh);
#pragma unroll
        for (int i = 0; i < 4; ++i) {
            int nt = wave * 4 + i;
            v16bf b = *(const v16bf*)(Wr_p + ((size_t)(kt * 16 + nt) * 32 + lane) * 16);
            acc0[i] = __builtin_amdgcn_wmma_f32_16x16x32_bf16(false, a0, false, b,
                                                              (short)0, acc0[i], false, false);
            acc1[i] = __builtin_amdgcn_wmma_f32_16x16x32_bf16(false, a1, false, b,
                                                              (short)0, acc1[i], false, false);
        }
    }

    // spill D tiles to LDS: lane col = nt*16+nlo, rows r + 8*kh (+16 for second tile)
#pragma unroll
    for (int i = 0; i < 4; ++i) {
        int col = (wave * 4 + i) * 16 + nlo;
#pragma unroll
        for (int r = 0; r < 8; ++r) {
            sbuf[(r + 8 * kh) * SBP_ + col]      = acc0[i][r];
            sbuf[(16 + r + 8 * kh) * SBP_ + col] = acc1[i][r];
        }
    }
    __syncthreads();

    // layernorm: 4 threads per row, 64 cols each
    const int row = t >> 2;
    const int sub = t & 3;
    const int c0  = sub * 64;
    const int gr  = (row < 16) ? (row0a + row) : (row0b + row - 16);
    const float* xr = x + (size_t)gr * D_ + c0;
    float* sp = sbuf + row * SBP_ + c0;
    float s = 0.0f, s2 = 0.0f;
#pragma unroll
    for (int c = 0; c < 64; c += 4) {
        float4 xv = *(const float4*)(xr + c);
        float4 bg = *(const float4*)(b_gnn + c0 + c);
        float4 u  = *(const float4*)(sp + c);
        u.x += xv.x + bg.x; u.y += xv.y + bg.y;
        u.z += xv.z + bg.z; u.w += xv.w + bg.w;
        s  += u.x + u.y + u.z + u.w;
        s2 += u.x * u.x + u.y * u.y + u.z * u.z + u.w * u.w;
        *(float4*)(sp + c) = u;
    }
    part[row][sub] = s; part2[row][sub] = s2;
    __syncthreads();
    if (sub == 0) {
        float S = 0.0f, S2 = 0.0f;
#pragma unroll
        for (int j = 0; j < 4; ++j) { S += part[row][j]; S2 += part2[row][j]; }
        float mu  = S * (1.0f / D_);
        float var = S2 * (1.0f / D_) - mu * mu;
        smu[row] = mu;
        srs[row] = rsqrtf(var + EPS_);
    }
    __syncthreads();
    const float mu = smu[row], rs = srs[row];
    float* ho = h + (size_t)gr * D_ + c0;
#pragma unroll
    for (int c = 0; c < 64; c += 4) {
        float4 g  = *(const float4*)(g1 + c0 + c);
        float4 be = *(const float4*)(beta1 + c0 + c);
        float4 u  = *(const float4*)(sp + c);
        float4 o;
        o.x = (u.x - mu) * rs * g.x + be.x;
        o.y = (u.y - mu) * rs * g.y + be.y;
        o.z = (u.z - mu) * rs * g.z + be.z;
        o.w = (u.w - mu) * rs * g.w + be.w;
        *(float4*)(ho + c) = o;
    }
}

// ---------- kernel 4: ff = relu(h@W1+b1)@W2 + b2 ; out = LN2(h + ff); 32 rows/block ----------
__global__ __launch_bounds__(128) void ffn_ln2(const float* __restrict__ h,
                                               const __bf16* __restrict__ W1p,
                                               const float* __restrict__ b1,
                                               const __bf16* __restrict__ W2p,
                                               const float* __restrict__ b2,
                                               const float* __restrict__ g2,
                                               const float* __restrict__ beta2,
                                               float* __restrict__ out, int N) {
    // st (bf16, 32 x STP_) and sbuf (f32, 32 x SBP_) alias: both 33,280 bytes.
    __shared__ __align__(16) char smem[32 * STP_ * sizeof(__bf16)];
    __bf16* st   = (__bf16*)smem;
    float*  sbuf = (float*)smem;
    __shared__ float part[32][4];
    __shared__ float part2[32][4];
    __shared__ float smu[32], srs[32];

    const int t    = threadIdx.x;
    const int wave = t >> 5;
    const int lane = t & 31;
    const int kh   = lane >> 4;
    const int nlo  = lane & 15;
    const int row0a = blockIdx.x * 32;
    const int row0b = min(row0a + 16, N - 16);
    const float* hA = h + (size_t)row0a * D_;
    const float* hB = h + (size_t)row0b * D_;

    // ---- stage 1: t = relu(h @ W1 + b1) -> bf16 in LDS (32 x 512) ----
    // two ntile-group passes to cap accumulator registers
#pragma unroll
    for (int ng = 0; ng < 2; ++ng) {
        v8f acc0[4] = {};
        v8f acc1[4] = {};
#pragma unroll
        for (int kt = 0; kt < 8; ++kt) {
            v16bf a0 = load_a_tile(hA, D_, nlo, kt * 32, kh);
            v16bf a1 = load_a_tile(hB, D_, nlo, kt * 32, kh);
#pragma unroll
            for (int i = 0; i < 4; ++i) {
                int nt = wave * 8 + ng * 4 + i;
                v16bf b = *(const v16bf*)(W1p + ((size_t)(kt * 32 + nt) * 32 + lane) * 16);
                acc0[i] = __builtin_amdgcn_wmma_f32_16x16x32_bf16(false, a0, false, b,
                                                                  (short)0, acc0[i], false, false);
                acc1[i] = __builtin_amdgcn_wmma_f32_16x16x32_bf16(false, a1, false, b,
                                                                  (short)0, acc1[i], false, false);
            }
        }
#pragma unroll
        for (int i = 0; i < 4; ++i) {
            int col = (wave * 8 + ng * 4 + i) * 16 + nlo;
            float bv = b1[col];
#pragma unroll
            for (int r = 0; r < 8; ++r) {
                st[(r + 8 * kh) * STP_ + col]      = f2bf(fmaxf(acc0[i][r] + bv, 0.0f));
                st[(16 + r + 8 * kh) * STP_ + col] = f2bf(fmaxf(acc1[i][r] + bv, 0.0f));
            }
        }
    }
    __syncthreads();

    // ---- stage 2: ff = t @ W2  (K = 512, A straight from bf16 LDS) ----
    v8f acc0[4] = {};
    v8f acc1[4] = {};
#pragma unroll
    for (int kt = 0; kt < 16; ++kt) {
        v16bf a0 = load_a_lds(st, STP_, nlo,      kt * 32, kh);
        v16bf a1 = load_a_lds(st, STP_, nlo + 16, kt * 32, kh);
#pragma unroll
        for (int i = 0; i < 4; ++i) {
            int nt = wave * 4 + i;
            v16bf b = *(const v16bf*)(W2p + ((size_t)(kt * 16 + nt) * 32 + lane) * 16);
            acc0[i] = __builtin_amdgcn_wmma_f32_16x16x32_bf16(false, a0, false, b,
                                                              (short)0, acc0[i], false, false);
            acc1[i] = __builtin_amdgcn_wmma_f32_16x16x32_bf16(false, a1, false, b,
                                                              (short)0, acc1[i], false, false);
        }
    }
    __syncthreads();   // all reads of st done; safe to reuse aliased LDS as sbuf
#pragma unroll
    for (int i = 0; i < 4; ++i) {
        int col = (wave * 4 + i) * 16 + nlo;
#pragma unroll
        for (int r = 0; r < 8; ++r) {
            sbuf[(r + 8 * kh) * SBP_ + col]      = acc0[i][r];
            sbuf[(16 + r + 8 * kh) * SBP_ + col] = acc1[i][r];
        }
    }
    __syncthreads();

    // ---- LN2 with residual h + bias b2: 4 threads per row, 64 cols each ----
    const int row = t >> 2;
    const int sub = t & 3;
    const int c0  = sub * 64;
    const int gr  = (row < 16) ? (row0a + row) : (row0b + row - 16);
    const float* hr = h + (size_t)gr * D_ + c0;
    float* sp = sbuf + row * SBP_ + c0;
    float s = 0.0f, s2 = 0.0f;
#pragma unroll
    for (int c = 0; c < 64; c += 4) {
        float4 hv = *(const float4*)(hr + c);
        float4 bg = *(const float4*)(b2 + c0 + c);
        float4 u  = *(const float4*)(sp + c);
        u.x += hv.x + bg.x; u.y += hv.y + bg.y;
        u.z += hv.z + bg.z; u.w += hv.w + bg.w;
        s  += u.x + u.y + u.z + u.w;
        s2 += u.x * u.x + u.y * u.y + u.z * u.z + u.w * u.w;
        *(float4*)(sp + c) = u;
    }
    part[row][sub] = s; part2[row][sub] = s2;
    __syncthreads();
    if (sub == 0) {
        float S = 0.0f, S2 = 0.0f;
#pragma unroll
        for (int j = 0; j < 4; ++j) { S += part[row][j]; S2 += part2[row][j]; }
        float mu  = S * (1.0f / D_);
        float var = S2 * (1.0f / D_) - mu * mu;
        smu[row] = mu;
        srs[row] = rsqrtf(var + EPS_);
    }
    __syncthreads();
    const float mu = smu[row], rs = srs[row];
    float* oo = out + (size_t)gr * D_ + c0;
#pragma unroll
    for (int c = 0; c < 64; c += 4) {
        float4 g  = *(const float4*)(g2 + c0 + c);
        float4 be = *(const float4*)(beta2 + c0 + c);
        float4 u  = *(const float4*)(sp + c);
        float4 o;
        o.x = (u.x - mu) * rs * g.x + be.x;
        o.y = (u.y - mu) * rs * g.y + be.y;
        o.z = (u.z - mu) * rs * g.z + be.z;
        o.w = (u.w - mu) * rs * g.w + be.w;
        *(float4*)(oo + c) = o;
    }
}

extern "C" void kernel_launch(void* const* d_in, const int* in_sizes, int n_in,
                              void* d_out, int out_size, void* d_ws, size_t ws_size,
                              hipStream_t stream) {
    const float*     x      = (const float*)d_in[0];
    const long long* ei     = (const long long*)d_in[1];
    const float*     W_nbr  = (const float*)d_in[2];
    const float*     W_root = (const float*)d_in[3];
    const float*     b_gnn  = (const float*)d_in[4];
    const float*     W1     = (const float*)d_in[5];
    const float*     b1     = (const float*)d_in[6];
    const float*     W2     = (const float*)d_in[7];
    const float*     b2     = (const float*)d_in[8];
    const float*     g1     = (const float*)d_in[9];
    const float*     beta1  = (const float*)d_in[10];
    const float*     g2     = (const float*)d_in[11];
    const float*     beta2  = (const float*)d_in[12];
    float*           outp   = (float*)d_out;

    const int N = in_sizes[0] / D_;
    const int E = in_sizes[1] / 2;

    char* ws = (char*)d_ws;
    const size_t aggBytes = (size_t)N * D_ * sizeof(float);
    float*   agg  = (float*)(ws);
    float*   hbuf = (float*)(ws + aggBytes);
    __bf16*  Wn_p = (__bf16*)(ws + 2 * aggBytes);
    __bf16*  Wr_p = Wn_p + (size_t)D_ * D_;
    __bf16*  W1_p = Wr_p + (size_t)D_ * D_;
    __bf16*  W2_p = W1_p + (size_t)D_ * DFF_;

    // 1) zero the aggregation buffer (atomics accumulate into it)
    {
        long long n = (long long)N * D_;
        zero_f32<<<2048, 256, 0, stream>>>(agg, n);
    }
    // 2) pack weights into WMMA bf16 B-tile layout
    {
        int t0 = (D_ / 32) * (D_ / 16) * 32;
        pack_weight<<<(t0 + 255) / 256, 256, 0, stream>>>(W_nbr,  Wn_p, D_,   D_);
        pack_weight<<<(t0 + 255) / 256, 256, 0, stream>>>(W_root, Wr_p, D_,   D_);
        int t1 = (D_ / 32) * (DFF_ / 16) * 32;
        pack_weight<<<(t1 + 255) / 256, 256, 0, stream>>>(W1, W1_p, D_,   DFF_);
        int t2 = (DFF_ / 32) * (D_ / 16) * 32;
        pack_weight<<<(t2 + 255) / 256, 256, 0, stream>>>(W2, W2_p, DFF_, D_);
    }
    // 3) edge scatter-add (L2-resident accumulator)
    {
        long long work = (long long)E * 64;
        int blocks = (int)((work + 255) / 256);
        scatter_add<<<blocks, 256, 0, stream>>>(x, ei, agg, E);
    }
    const int blocks32 = (N + 31) / 32;
    // 4) GNN transform + LN1
    gnn_ln1<<<blocks32, 128, 0, stream>>>(x, agg, Wn_p, Wr_p, b_gnn, g1, beta1, hbuf, N);
    // 5) fused FFN + LN2
    ffn_ln2<<<blocks32, 128, 0, stream>>>(hbuf, W1_p, b1, W2_p, b2, g2, beta2, outp, N);
}
